// Net_89163521065694
// MI455X (gfx1250) — matrix operand order repaired and checked by reading, not compiled, not run on hardware
//
#include <hip/hip_runtime.h>
#include <hip/hip_bf16.h>

typedef __attribute__((ext_vector_type(16))) _Float16 v16h;
typedef __attribute__((ext_vector_type(8)))  float    v8f;

#define NTHREADS 256

// ---------------- WMMA fragment loaders (CDNA5 wave32 layouts) ----------------
// A-matrix 16x32 f16: lanes 0-15 -> M=0-15 (half=0), lanes 16-31 -> M=0-15 (half=1)
// element e<8:  K = half*8 + e ; e>=8: K = 16 + half*8 + (e-8)
__device__ __forceinline__ void load_a_frag(const _Float16* __restrict__ src,
                                            int mBase, int kBase, int lane, v16h& A) {
  const int half = lane >> 4;
  const int m = mBase + (lane & 15);
  const _Float16* p0 = src + m * 64 + kBase + half * 8;
#pragma unroll
  for (int e = 0; e < 8; ++e) { A[e] = p0[e]; A[e + 8] = p0[e + 16]; }
}

// B-matrix 32x16 f16: lane -> N = lane%16 ; element e<8: K = half*8 + e ; e>=8: K = 16 + half*8 + (e-8)
__device__ __forceinline__ void load_b_frag(const _Float16* __restrict__ src,
                                            int kBase, int nBase, int lane, v16h& Bf) {
  const int half = lane >> 4;
  const _Float16* p = src + (kBase + half * 8) * 64 + nBase + (lane & 15);
#pragma unroll
  for (int e = 0; e < 8; ++e) { Bf[e] = p[e * 64]; Bf[e + 8] = p[(e + 16) * 64]; }
}

// ---------------- structured matvec: sY = Dmat @ x  (x given as f16 in sXh) ----
// Phase 1: Z_a[k][j] = sum_l Xh[k][l] * TA2T[a][l][j]      (5 matmuls 64x64x64)
// Phase 2: Y[i][j]   = sum_a sum_k U_a[i][k] * Z_a[k][j]   (5 matmuls, accumulated)
__device__ __forceinline__ void matvec(const _Float16* sXh, const _Float16* sTA2T,
                                       const _Float16* sU, _Float16* sZ, float* sY,
                                       int tid) {
  __syncthreads();
  const int wave = tid >> 5;
  const int lane = tid & 31;
  // Phase 1: 80 tile-jobs (5 a-values x 16 output tiles), 10 per wave (uniform)
  for (int job = wave; job < 80; job += 8) {
    const int a  = job / 16;
    const int t  = job % 16;
    const int mB = (t >> 2) << 4;
    const int nB = (t & 3) << 4;
    v8f acc = {};
#pragma unroll
    for (int kc = 0; kc < 2; ++kc) {
      v16h Af, Bf;
      load_a_frag(sXh, mB, kc * 32, lane, Af);
      load_b_frag(sTA2T + a * 4096, kc * 32, nB, lane, Bf);
      acc = __builtin_amdgcn_wmma_f32_16x16x32_f16(false, Af, false, Bf,
                                                   (short)0, acc, false, false);
    }
    const int n  = nB + (lane & 15);
    const int mo = mB + ((lane >> 4) << 3);
#pragma unroll
    for (int v = 0; v < 8; ++v)
      sZ[a * 4096 + (mo + v) * 64 + n] = (_Float16)acc[v];
  }
  __syncthreads();
  // Phase 2: 16 output tiles, 2 per wave, 10 WMMAs each (uniform)
  for (int t = wave; t < 16; t += 8) {
    const int mB = (t >> 2) << 4;
    const int nB = (t & 3) << 4;
    v8f acc = {};
#pragma unroll
    for (int a = 0; a < 5; ++a) {
#pragma unroll
      for (int kc = 0; kc < 2; ++kc) {
        v16h Af, Bf;
        load_a_frag(sU + a * 4096, mB, kc * 32, lane, Af);
        load_b_frag(sZ + a * 4096, kc * 32, nB, lane, Bf);
        acc = __builtin_amdgcn_wmma_f32_16x16x32_f16(false, Af, false, Bf,
                                                     (short)0, acc, false, false);
      }
    }
    const int n  = nB + (lane & 15);
    const int mo = mB + ((lane >> 4) << 3);
#pragma unroll
    for (int v = 0; v < 8; ++v)
      sY[(mo + v) * 64 + n] = acc[v];
  }
  __syncthreads();
}

// ---------------- masked Sinkhorn on a 64x64 f32 LDS matrix --------------------
__device__ __forceinline__ void sinkhorn(float* S, float* tmp, int tid, int iters) {
  for (int it = 0; it < iters; ++it) {
    if (tid < 64) { float s = 0.f; const float* r = S + tid * 64;
      for (int j = 0; j < 64; ++j) s += r[j]; tmp[tid] = s; }
    __syncthreads();
    for (int idx = tid; idx < 4096; idx += NTHREADS) {
      const int i = idx >> 6; if (i < 63) S[idx] /= tmp[i];
    }
    __syncthreads();
    if (tid < 64) { float s = 0.f;
      for (int i = 0; i < 64; ++i) s += S[i * 64 + tid]; tmp[tid] = s; }
    __syncthreads();
    for (int idx = tid; idx < 4096; idx += NTHREADS) {
      const int j = idx & 63; if (j < 63) S[idx] /= tmp[j];
    }
    __syncthreads();
  }
}

__global__ __launch_bounds__(NTHREADS)
void ged_fw_kernel(const float* __restrict__ node_w, const float* __restrict__ edge_w,
                   const int* __restrict__ A1, const int* __restrict__ A2,
                   const int* __restrict__ l1, const int* __restrict__ l2,
                   float* __restrict__ ged_out) {
  __shared__ _Float16 sTA2T[5 * 4096];   // T[a, A2[j,l]] stored [a][l][j]
  __shared__ _Float16 sU[5 * 4096];      // one-hot of A1 stored [a][i][k]
  __shared__ _Float16 sZ[5 * 4096];      // intermediate Z_a
  __shared__ _Float16 sXh[4096];         // f16 copy of matvec input
  __shared__ float sX[4096], sC[4096], sG[4096], sS[4096], sY[4096];
  __shared__ float sNode[64];            // 8x8 node cost table
  __shared__ float sT[25];               // 5x5 edge cost table
  __shared__ _Float16 sTh[25];
  __shared__ float sRed[NTHREADS];
  __shared__ float sScal[4];

  const int tid = threadIdx.x;
  const int b   = blockIdx.x;

  // ---- tiny cost tables (one thread) ----
  if (tid == 0) {
    const float nid = fmaxf(node_w[28], 0.f);
    sScal[0] = nid;
    int p = 0;
    for (int r = 0; r < 8; ++r) {
      sNode[r * 8 + r] = 0.f;
      for (int c = r + 1; c < 8; ++c) {
        const float v = fmaxf(node_w[p++], 0.f);
        sNode[r * 8 + c] = v; sNode[c * 8 + r] = v;
      }
    }
    float E[16];
    p = 0;
    for (int r = 0; r < 4; ++r) {
      E[r * 4 + r] = 0.f;
      for (int c = r + 1; c < 4; ++c) {
        const float v = fmaxf(edge_w[p++], 0.f);
        E[r * 4 + c] = v; E[c * 4 + r] = v;
      }
    }
    const float eid = fmaxf(edge_w[6], 0.f);
    sT[0] = 0.f;
    for (int k = 1; k < 5; ++k) { sT[k] = eid; sT[k * 5] = eid; }
    for (int r = 1; r < 5; ++r)
      for (int c = 1; c < 5; ++c) sT[r * 5 + c] = 2.f * E[(r - 1) * 4 + (c - 1)];
    for (int i = 0; i < 25; ++i) sTh[i] = (_Float16)sT[i];
  }
  __syncthreads();

  const int* A1b = A1 + b * 63 * 63;
  const int* A2b = A2 + b * 63 * 63;
  const int* l1b = l1 + b * 63;
  const int* l2b = l2 + b * 63;
  const float nid = sScal[0];

  // ---- build factor matrices, c, and exp(-c) ----
  for (int idx = tid; idx < 4096; idx += NTHREADS) {
    const int r = idx >> 6, c = idx & 63;
    const bool in = (r < 63) && (c < 63);
    const int a2 = in ? A2b[r * 63 + c] : 0;   // A1/A2 symmetric: [r][c]==[c][r]
    const int a1 = in ? A1b[r * 63 + c] : 0;
#pragma unroll
    for (int a = 0; a < 5; ++a) {
      sTA2T[a * 4096 + idx] = sTh[a * 5 + a2];
      sU[a * 4096 + idx]    = (a1 == a) ? (_Float16)1.0f : (_Float16)0.0f;
    }
    float cv;
    if (in) cv = sNode[l1b[r] * 8 + l2b[c]];
    else if (r == 63 && c == 63) cv = 0.f;
    else cv = nid;
    sC[idx] = cv;
    sS[idx] = __expf(-cv);
  }
  __syncthreads();

  // ---- x0 = sinkhorn(exp(-c), 10) ----
  sinkhorn(sS, sRed, tid, 10);
  for (int idx = tid; idx < 4096; idx += NTHREADS) sX[idx] = sS[idx];

  // ---- Frank-Wolfe loop ----
  for (int iter = 0; iter < 15; ++iter) {
    for (int idx = tid; idx < 4096; idx += NTHREADS) sXh[idx] = (_Float16)sX[idx];
    matvec(sXh, sTA2T, sU, sZ, sY, tid);                 // sY = D @ x
    for (int idx = tid; idx < 4096; idx += NTHREADS) {
      const float g = sC[idx] + sY[idx];
      sG[idx] = g;
      sS[idx] = __expf(-g);
    }
    __syncthreads();
    sinkhorn(sS, sRed, tid, 5);                          // sS = b

    float pn = 0.f;
    for (int idx = tid; idx < 4096; idx += NTHREADS) {
      const float dv = sS[idx] - sX[idx];                // d = b - x
      sXh[idx] = (_Float16)dv;
      pn += dv * sG[idx];
    }
    sRed[tid] = pn;
    __syncthreads();
    if (tid == 0) { float s = 0.f; for (int k = 0; k < NTHREADS; ++k) s += sRed[k]; sScal[1] = s; }

    matvec(sXh, sTA2T, sU, sZ, sY, tid);                 // sY = D @ d

    float pd = 0.f;
    for (int idx = tid; idx < 4096; idx += NTHREADS) {
      const float dv = sS[idx] - sX[idx];
      pd += dv * sY[idx];
    }
    sRed[tid] = pd;
    __syncthreads();
    if (tid == 0) {
      float den = 0.f; for (int k = 0; k < NTHREADS; ++k) den += sRed[k];
      const float num = sScal[1];
      float tstep;
      if (den > 0.f) tstep = fminf(fmaxf(-num / den, 0.f), 1.f);
      else           tstep = (num < 0.f) ? 1.f : 0.f;
      sScal[3] = tstep;
    }
    __syncthreads();
    const float tstep = sScal[3];
    for (int idx = tid; idx < 4096; idx += NTHREADS)
      sX[idx] += tstep * (sS[idx] - sX[idx]);
    __syncthreads();
  }

  // ---- ged = 0.5 x^T D x + c . x ----
  for (int idx = tid; idx < 4096; idx += NTHREADS) sXh[idx] = (_Float16)sX[idx];
  matvec(sXh, sTA2T, sU, sZ, sY, tid);
  float pg = 0.f;
  for (int idx = tid; idx < 4096; idx += NTHREADS)
    pg += (0.5f * sY[idx] + sC[idx]) * sX[idx];
  sRed[tid] = pg;
  __syncthreads();
  if (tid == 0) {
    float s = 0.f; for (int k = 0; k < NTHREADS; ++k) s += sRed[k];
    ged_out[b] = s;
  }
}

__global__ void ged_norm_kernel(const float* __restrict__ ged, float* __restrict__ out) {
  if (threadIdx.x == 0) {
    float mn = ged[0], mx = ged[0];
    for (int i = 1; i < 16; ++i) { mn = fminf(mn, ged[i]); mx = fmaxf(mx, ged[i]); }
    const float inv = 1.f / (mx - mn);
    for (int i = 0; i < 16; ++i) out[i] = (ged[i] - mn) * inv;
  }
}

extern "C" void kernel_launch(void* const* d_in, const int* in_sizes, int n_in,
                              void* d_out, int out_size, void* d_ws, size_t ws_size,
                              hipStream_t stream) {
  const float* node_w = (const float*)d_in[0];
  const float* edge_w = (const float*)d_in[1];
  const int*   A1     = (const int*)d_in[2];
  const int*   A2     = (const int*)d_in[3];
  const int*   l1     = (const int*)d_in[4];
  const int*   l2     = (const int*)d_in[5];
  float* ged = (float*)d_ws;
  float* out = (float*)d_out;

  hipLaunchKernelGGL(ged_fw_kernel, dim3(16), dim3(NTHREADS), 0, stream,
                     node_w, edge_w, A1, A2, l1, l2, ged);
  hipLaunchKernelGGL(ged_norm_kernel, dim3(1), dim3(32), 0, stream, ged, out);
}
